// SRN_46763603919230
// MI455X (gfx1250) — compile-verified, matrix operand order checked by reading, not run on hardware
//
#include <hip/hip_runtime.h>

// Problem constants (match the reference)
#define V_  50000
#define D_  128
#define H_  256
#define C_  128
#define R_  7
#define B_  1024
#define T_  128
#define KH  (D_ + H_)                 // 384, K of the recurrence GEMM
#define KC_IH   (KH / 32)             // 12 K-chunks of 32 for Wih/Wio
#define KC_CPR  ((2 * H_) / 32)       // 16 K-chunks for Wcpr
#define NT_H    (H_ / 16)             // 16 column tiles of H
#define NT_C    (C_ / 16)             // 8 column tiles of C

typedef __attribute__((ext_vector_type(16))) __bf16          v16bf;
typedef __attribute__((ext_vector_type(8)))  float           v8f;
typedef __attribute__((ext_vector_type(16))) unsigned short  v16us;
typedef __attribute__((ext_vector_type(8)))  unsigned short  v8us;

__device__ __forceinline__ unsigned short f32_to_bf16(float f) {
  unsigned int u = __builtin_bit_cast(unsigned int, f);
  u += 0x7FFFu + ((u >> 16) & 1u);          // round-to-nearest-even
  return (unsigned short)(u >> 16);
}

// K index inside a 32-wide chunk for element j of a lane's 16-element
// bf16 WMMA fragment (mirrors the ISA 16-bit A/B striping: lanes 0-15
// hold K {0..7,16..23}, lanes 16-31 hold K {8..15,24..31}).
__device__ __forceinline__ int kmap(int lane, int j) {
  return ((j < 8) ? j : (j + 8)) + ((lane >= 16) ? 8 : 0);
}

// B fragment: 16 contiguous bf16 per lane (32B) from prepped weight array.
__device__ __forceinline__ v16bf load_b_frag(const unsigned short* p) {
  v16us w = *(const v16us*)p;
  return __builtin_bit_cast(v16bf, w);
}

// A fragment from an LDS-staged row: two contiguous 8-element (16B) chunks.
__device__ __forceinline__ v16bf load_a_frag(const unsigned short* combRow, int kbase) {
  v8us lo = *(const v8us*)(combRow + kbase);
  v8us hi = *(const v8us*)(combRow + kbase + 16);
  v16us a;
#pragma unroll
  for (int j = 0; j < 8; ++j) { a[j] = lo[j]; a[j + 8] = hi[j]; }
  return __builtin_bit_cast(v16bf, a);
}

// ---------------------------------------------------------------------------
// Kernel 1: convert fp32 weights into bf16 B-fragment layout in workspace.
// wih_f/wio_f: [NT_H][KC_IH][32 lanes][16 elems], wcpr_f: [NT_C][KC_CPR][32][16]
// ---------------------------------------------------------------------------
__global__ __launch_bounds__(256)
void srn_prep_kernel(const float* __restrict__ Wih,
                     const float* __restrict__ Wio,
                     const float* __restrict__ Wcpr,
                     unsigned short* __restrict__ wih_f,
                     unsigned short* __restrict__ wio_f,
                     unsigned short* __restrict__ wcpr_f) {
  const int NIH  = NT_H * KC_IH * 32 * 16;    // 98304
  const int NCPR = NT_C * KC_CPR * 32 * 16;   // 65536
  int gid = blockIdx.x * blockDim.x + threadIdx.x;
  if (gid < 2 * NIH) {
    int e = (gid < NIH) ? gid : (gid - NIH);
    int j = e & 15, lane = (e >> 4) & 31, r2 = e >> 9;
    int kc = r2 % KC_IH, nt = r2 / KC_IH;
    int n = nt * 16 + (lane & 15);
    int k = kc * 32 + kmap(lane, j);
    if (gid < NIH) wih_f[e] = f32_to_bf16(Wih[n * KH + k]);
    else           wio_f[e] = f32_to_bf16(Wio[n * KH + k]);
  } else if (gid < 2 * NIH + NCPR) {
    int e = gid - 2 * NIH;
    int j = e & 15, lane = (e >> 4) & 31, r2 = e >> 9;
    int kc = r2 % KC_CPR, nt = r2 / KC_CPR;
    int n = nt * 16 + (lane & 15);
    int k = kc * 32 + kmap(lane, j);
    wcpr_f[e] = f32_to_bf16(Wcpr[n * (2 * H_) + k]);
  }
}

// ---------------------------------------------------------------------------
// Kernel 2: linear RNN recurrence. One block = one (16-row batch tile, side).
// 16 waves; wave w owns H-columns [16w,16w+16). Wih fragments live in 96
// VGPRs for all steps; Wio is swapped in only for the final i2o step.
// ---------------------------------------------------------------------------
__global__ __launch_bounds__(512)
void srn_rnn_kernel(const int* __restrict__ left_ids,
                    const int* __restrict__ right_ids,
                    const float* __restrict__ emb,
                    const float* __restrict__ bih,
                    const float* __restrict__ bio,
                    const unsigned short* __restrict__ wih_f,
                    const unsigned short* __restrict__ wio_f,
                    float* __restrict__ rnn_out) {
  __shared__ __attribute__((aligned(16))) unsigned short comb[16][KH]; // 12 KB

  const int tid    = threadIdx.x;
  const int lane   = tid & 31;
  const int wave   = tid >> 5;            // 0..15 : H column tile
  const int col    = lane & 15;           // N within tile
  const int hiHalf = (lane >= 16) ? 1 : 0;
  const int koff   = hiHalf * 8;
  const int bt     = blockIdx.x;
  const int side   = blockIdx.y;
  const int* ids   = side ? right_ids : left_ids;
  const int b0     = bt * 16;

  const float bih_v = bih[wave * 16 + col];
  const float bio_v = bio[wave * 16 + col];

  // Wih B-fragments resident in registers (reused for T-1 steps).
  v16bf wreg[KC_IH];
#pragma unroll
  for (int kc = 0; kc < KC_IH; ++kc)
    wreg[kc] = load_b_frag(wih_f + ((wave * KC_IH + kc) * 32 + lane) * 16);

  v8f h;
#pragma unroll
  for (int r = 0; r < 8; ++r) h[r] = 0.0f;

  for (int t = 0; t < T_; ++t) {
    // Stage x_t = emb[ids[:, t]] (16 x 128) as bf16 into comb[:, 0:128].
    for (int e = tid; e < 16 * D_; e += 512) {
      int row = e >> 7;
      int d   = e & (D_ - 1);
      int id  = ids[(b0 + row) * T_ + t];
      comb[row][d] = f32_to_bf16(emb[id * D_ + d]);
      if (t + 1 < T_) {   // prefetch next step's embedding row into caches
        int id2 = ids[(b0 + row) * T_ + t + 1];
        __builtin_prefetch(&emb[id2 * D_ + d], 0, 1);
      }
    }
    // Stage h_t (bf16) into comb[:, 128:384] from this wave's accumulator.
#pragma unroll
    for (int r = 0; r < 8; ++r)
      comb[r + hiHalf * 8][D_ + wave * 16 + col] = f32_to_bf16(h[r]);
    __syncthreads();

    const bool last = (t == T_ - 1);
    if (last) {  // final step uses the i2o weights instead of the recurrence
#pragma unroll
      for (int kc = 0; kc < KC_IH; ++kc)
        wreg[kc] = load_b_frag(wio_f + ((wave * KC_IH + kc) * 32 + lane) * 16);
    }
    const float bias = last ? bio_v : bih_v;
    v8f acc;
#pragma unroll
    for (int r = 0; r < 8; ++r) acc[r] = bias;
#pragma unroll
    for (int kc = 0; kc < KC_IH; ++kc) {
      v16bf a = load_a_frag(&comb[col][0], kc * 32 + koff);
      acc = __builtin_amdgcn_wmma_f32_16x16x32_bf16(
          false, a, false, wreg[kc], (short)0, acc, false, false);
    }
    __syncthreads();
    h = acc;   // h_{t+1}; on the last step this is the i2o output
  }

  float* outp = rnn_out + (side * B_ + b0) * H_ + wave * 16 + col;
#pragma unroll
  for (int r = 0; r < 8; ++r)
    outp[(r + hiHalf * 8) * H_] = h[r];
}

// ---------------------------------------------------------------------------
// Kernel 3: head. concat -> Wcpr (WMMA) -> LeakyReLU -> Wsm (R=7, VALU)
//           -> log_softmax. One block = one 16-row batch tile, 8 waves.
// ---------------------------------------------------------------------------
__global__ __launch_bounds__(256)
void srn_head_kernel(const float* __restrict__ rnn_out,
                     const unsigned short* __restrict__ wcpr_f,
                     const float* __restrict__ bcpr,
                     const float* __restrict__ Wsm,
                     const float* __restrict__ bsm,
                     float* __restrict__ out) {
  __shared__ __attribute__((aligned(16))) unsigned short concat[16][2 * H_]; // 16 KB
  __shared__ float act[16][C_];                                              // 8 KB
  __shared__ float lg[16][8];
  __shared__ float lse[16];

  const int tid    = threadIdx.x;
  const int lane   = tid & 31;
  const int wave   = tid >> 5;            // 0..7 : C column tile
  const int col    = lane & 15;
  const int hiHalf = (lane >= 16) ? 1 : 0;
  const int koff   = hiHalf * 8;
  const int b0     = blockIdx.x * 16;

  // Stage concat = [left_out, right_out] (16 x 512) as bf16.
  for (int e = tid; e < 16 * 2 * H_; e += 256) {
    int row = e >> 9;
    int c   = e & (2 * H_ - 1);
    float v = (c < H_) ? rnn_out[(b0 + row) * H_ + c]
                       : rnn_out[(B_ + b0 + row) * H_ + (c - H_)];
    concat[row][c] = f32_to_bf16(v);
  }
  __syncthreads();

  float bias = bcpr[wave * 16 + col];
  v8f acc;
#pragma unroll
  for (int r = 0; r < 8; ++r) acc[r] = bias;
#pragma unroll
  for (int kc = 0; kc < KC_CPR; ++kc) {
    v16bf a = load_a_frag(&concat[col][0], kc * 32 + koff);
    v16bf b = load_b_frag(wcpr_f + ((wave * KC_CPR + kc) * 32 + lane) * 16);
    acc = __builtin_amdgcn_wmma_f32_16x16x32_bf16(
        false, a, false, b, (short)0, acc, false, false);
  }
  // LeakyReLU, park activations in LDS as fp32.
#pragma unroll
  for (int r = 0; r < 8; ++r) {
    float v = acc[r];
    v = (v >= 0.0f) ? v : 0.01f * v;
    act[r + hiHalf * 8][wave * 16 + col] = v;
  }
  __syncthreads();

  // logits[16][7] = act @ Wsm^T + bsm  (tiny: plain VALU dot products)
  if (tid < 16 * R_) {
    int row = tid / R_;
    int r   = tid % R_;
    float s = bsm[r];
    for (int c = 0; c < C_; ++c) s += act[row][c] * Wsm[r * C_ + c];
    lg[row][r] = s;
  }
  __syncthreads();
  if (tid < 16) {
    float m = lg[tid][0];
#pragma unroll
    for (int r = 1; r < R_; ++r) m = fmaxf(m, lg[tid][r]);
    float sum = 0.0f;
#pragma unroll
    for (int r = 0; r < R_; ++r) sum += __expf(lg[tid][r] - m);
    lse[tid] = m + __logf(sum);
  }
  __syncthreads();
  if (tid < 16 * R_) {
    int row = tid / R_;
    int r   = tid % R_;
    out[(b0 + row) * R_ + r] = lg[row][r] - lse[row];
  }
}

// ---------------------------------------------------------------------------
extern "C" void kernel_launch(void* const* d_in, const int* in_sizes, int n_in,
                              void* d_out, int out_size, void* d_ws, size_t ws_size,
                              hipStream_t stream) {
  (void)in_sizes; (void)n_in; (void)out_size; (void)ws_size;
  const int*   left  = (const int*)d_in[0];
  const int*   right = (const int*)d_in[1];
  const float* emb   = (const float*)d_in[2];
  const float* Wih   = (const float*)d_in[3];
  const float* bih   = (const float*)d_in[4];
  const float* Wio   = (const float*)d_in[5];
  const float* bio   = (const float*)d_in[6];
  const float* Wcpr  = (const float*)d_in[7];
  const float* bcpr  = (const float*)d_in[8];
  const float* Wsm   = (const float*)d_in[9];
  const float* bsm   = (const float*)d_in[10];

  char* ws = (char*)d_ws;
  unsigned short* wih_f  = (unsigned short*)(ws + 0);        // 196608 B
  unsigned short* wio_f  = (unsigned short*)(ws + 196608);   // 196608 B
  unsigned short* wcpr_f = (unsigned short*)(ws + 393216);   // 131072 B
  float*          rnnout = (float*)(ws + 524288);            // 2 MB (2*B*H f32)

  srn_prep_kernel<<<1024, 256, 0, stream>>>(Wih, Wio, Wcpr, wih_f, wio_f, wcpr_f);
  srn_rnn_kernel<<<dim3(B_ / 16, 2), 512, 0, stream>>>(
      left, right, emb, bih, bio, wih_f, wio_f, rnnout);
  srn_head_kernel<<<B_ / 16, 256, 0, stream>>>(
      rnnout, wcpr_f, bcpr, Wsm, bsm, (float*)d_out);
}
